// SelfAttention_13718125543934
// MI455X (gfx1250) — matrix operand order repaired and checked by reading, not compiled, not run on hardware
//
#include <hip/hip_runtime.h>
#include <math.h>

typedef __bf16 bf16;
typedef __attribute__((ext_vector_type(16))) __bf16 bf16x16;
typedef __attribute__((ext_vector_type(8)))  float   f32x8;
typedef __attribute__((ext_vector_type(4)))  int     v4i;

#define B_  16
#define C_  512
#define N_  1024
#define H_  8
#define D_  64

// ---------------------------------------------------------------------------
// gfx1250 async global->LDS copy (ASYNCcnt path), with safe fallback.
// Builtin signature (from hipcc diagnostic): (v4i AS1*, v4i AS3*, Ii, Ii).
// ---------------------------------------------------------------------------
#if defined(__has_builtin)
#if __has_builtin(__builtin_amdgcn_global_load_async_to_lds_b128) && \
    __has_builtin(__builtin_amdgcn_s_wait_asynccnt)
#define USE_ASYNC_LDS 1
#endif
#endif
#ifndef USE_ASYNC_LDS
#define USE_ASYNC_LDS 0
#endif

__device__ __forceinline__ void copy16_to_lds(bf16* lds, const bf16* g) {
#if USE_ASYNC_LDS
  __builtin_amdgcn_global_load_async_to_lds_b128(
      (__attribute__((address_space(1))) v4i*)g,
      (__attribute__((address_space(3))) v4i*)lds, 0, 0);
#else
  *(uint4*)lds = *(const uint4*)g;
#endif
}

__device__ __forceinline__ void wait_lds_copies() {
#if USE_ASYNC_LDS
  __builtin_amdgcn_s_wait_asynccnt(0);
#endif
}

struct U8 { unsigned int u[8]; };

// A-fragment (16x32 bf16, M x K), source row-major [M][K] with leading dim ld.
// ISA layout: lane m=lane&15, half=lane>>4; VGPR 0..3 hold K=half*8+{0..7},
// VGPR 4..7 hold K=16+half*8+{0..7}  -> two contiguous 16B chunks per lane.
__device__ __forceinline__ bf16x16 frag_a(const bf16* base, int ld) {
  int lane = threadIdx.x & 31;
  int m = lane & 15, h = lane >> 4;
  const uint4* r = (const uint4*)(base + (size_t)m * ld);
  uint4 lo = r[h];
  uint4 hi = r[2 + h];
  U8 f = { lo.x, lo.y, lo.z, lo.w, hi.x, hi.y, hi.z, hi.w };
  return __builtin_bit_cast(bf16x16, f);
}

// B-fragment (32x16 bf16, K x N), source is B^T row-major [N][K], leading dim ld.
// ISA layout: lane n=lane&15, half=lane>>4; VGPR 0..3 hold K=half*8+{0..7},
// VGPR 4..7 hold K=16+half*8+{0..7}  -> two contiguous 16B chunks per lane.
__device__ __forceinline__ bf16x16 frag_b(const bf16* base, int ld) {
  int lane = threadIdx.x & 31;
  int n = lane & 15, h = lane >> 4;
  const uint4* r = (const uint4*)(base + (size_t)n * ld);
  uint4 lo = r[h];
  uint4 hi = r[2 + h];
  U8 f = { lo.x, lo.y, lo.z, lo.w, hi.x, hi.y, hi.z, hi.w };
  return __builtin_bit_cast(bf16x16, f);
}

__device__ __forceinline__ f32x8 wmma_bf16(bf16x16 a, bf16x16 b, f32x8 c) {
  return __builtin_amdgcn_wmma_f32_16x16x32_bf16(false, a, false, b, (short)0, c,
                                                 false, false);
}

// ---------------------------------------------------------------------------
// Kernel 1: weights f32 -> bf16
// ---------------------------------------------------------------------------
__global__ __launch_bounds__(256) void cvt_weights(const float* __restrict__ wq,
                                                   const float* __restrict__ wp,
                                                   bf16* __restrict__ wqb,
                                                   bf16* __restrict__ wpb) {
  int i = blockIdx.x * 256 + threadIdx.x;
  const int nq = 3 * C_ * C_;  // 786432
  if (i < nq) wqb[i] = (bf16)wq[i];
  else        wpb[i - nq] = (bf16)wp[i - nq];
}

// ---------------------------------------------------------------------------
// Kernel 2: GroupNorm32 (f32 stats) -> x_norm^T [b][n][c] in bf16
// One block per (b, group); group = 16 consecutive channels * 1024 px = 16384 f32.
// ---------------------------------------------------------------------------
__global__ __launch_bounds__(256) void groupnorm_kernel(
    const float* __restrict__ x, const float* __restrict__ gamma,
    const float* __restrict__ beta, bf16* __restrict__ xn_t) {
  int b = blockIdx.x >> 5, g = blockIdx.x & 31;
  int cstart = g * 16;
  const float* xb = x + ((size_t)b * C_ + cstart) * N_;
  float sum = 0.f, ss = 0.f;
  for (int i = threadIdx.x; i < 16 * N_; i += 256) {
    float v = xb[i];
    sum += v;
    ss += v * v;
  }
#pragma unroll
  for (int off = 16; off >= 1; off >>= 1) {
    sum += __shfl_xor(sum, off, 32);
    ss += __shfl_xor(ss, off, 32);
  }
  __shared__ float red[2][8];
  int wave = threadIdx.x >> 5, lane = threadIdx.x & 31;
  if (lane == 0) { red[0][wave] = sum; red[1][wave] = ss; }
  __syncthreads();
  sum = 0.f; ss = 0.f;
#pragma unroll
  for (int w = 0; w < 8; w++) { sum += red[0][w]; ss += red[1][w]; }
  float mean = sum * (1.f / 16384.f);
  float var = ss * (1.f / 16384.f) - mean * mean;
  float rstd = rsqrtf(var + 1e-6f);
  for (int i = threadIdx.x; i < 16 * N_; i += 256) {
    int cl = i >> 10, p = i & 1023, c = cstart + cl;
    float v = (xb[i] - mean) * rstd * gamma[c] + beta[c];
    xn_t[((size_t)b * N_ + p) * C_ + c] = (bf16)v;
  }
}

// ---------------------------------------------------------------------------
// Kernel 3/5: WMMA GEMM  C[M][N] = A[M][K=512] * Bt[N][K=512]^T  (per batch b)
// Block tile 128(M) x 64(N); 8 waves, each 32x32; K staged 32 at a time in LDS
// via gfx1250 async global->LDS copies.
// mode 0: QKV epilogue (bias, split to q[b,h,n,d], k[b,h,n,d], v[b,h,d,n])
// mode 1: proj epilogue (bias + residual into f32 out[b,c,n])
// ---------------------------------------------------------------------------
__global__ __launch_bounds__(256) void gemm_wmma(
    const bf16* __restrict__ A, const bf16* __restrict__ Bt,
    const float* __restrict__ bias, bf16* __restrict__ qd,
    bf16* __restrict__ kd, bf16* __restrict__ vd, float* __restrict__ outd,
    const float* __restrict__ xres, int mode) {
  __shared__ __align__(16) bf16 sA[128 * 32];
  __shared__ __align__(16) bf16 sB[64 * 32];
  int b = blockIdx.z, M0 = blockIdx.y * 128, N0 = blockIdx.x * 64;
  int tid = threadIdx.x, wave = tid >> 5, lane = tid & 31;
  int wm = wave >> 1, wn = wave & 1;
  f32x8 z = {0.f, 0.f, 0.f, 0.f, 0.f, 0.f, 0.f, 0.f};
  f32x8 acc[2][2] = {{z, z}, {z, z}};
  const bf16* Bb = Bt + (size_t)b * N_ * C_;
  // per-thread source/dest for the cooperative tile copies
  int rowA0 = (tid + 0)   >> 2, c4A0 = (tid + 0)   & 3;
  int rowA1 = (tid + 256) >> 2, c4A1 = (tid + 256) & 3;
  int rowB  = tid >> 2,         c4B  = tid & 3;
  for (int k0 = 0; k0 < C_; k0 += 32) {
    copy16_to_lds(sA + rowA0 * 32 + c4A0 * 8,
                  A + (size_t)(M0 + rowA0) * C_ + k0 + c4A0 * 8);
    copy16_to_lds(sA + rowA1 * 32 + c4A1 * 8,
                  A + (size_t)(M0 + rowA1) * C_ + k0 + c4A1 * 8);
    copy16_to_lds(sB + rowB * 32 + c4B * 8,
                  Bb + (size_t)(N0 + rowB) * C_ + k0 + c4B * 8);
    if (k0 + 32 < C_) {  // prefetch next K tile into caches
      __builtin_prefetch(A + (size_t)(M0 + rowA0) * C_ + k0 + 32 + c4A0 * 8, 0, 1);
      __builtin_prefetch(Bb + (size_t)(N0 + rowB) * C_ + k0 + 32 + c4B * 8, 0, 1);
    }
    wait_lds_copies();
    __syncthreads();
    bf16x16 af0 = frag_a(sA + (wm * 32 + 0) * 32, 32);
    bf16x16 af1 = frag_a(sA + (wm * 32 + 16) * 32, 32);
    bf16x16 bf0 = frag_b(sB + (wn * 32 + 0) * 32, 32);
    bf16x16 bf1 = frag_b(sB + (wn * 32 + 16) * 32, 32);
    acc[0][0] = wmma_bf16(af0, bf0, acc[0][0]);
    acc[0][1] = wmma_bf16(af0, bf1, acc[0][1]);
    acc[1][0] = wmma_bf16(af1, bf0, acc[1][0]);
    acc[1][1] = wmma_bf16(af1, bf1, acc[1][1]);
    __syncthreads();
  }
  int nn = lane & 15, hf = lane >> 4;
#pragma unroll
  for (int i = 0; i < 2; i++)
#pragma unroll
    for (int j = 0; j < 2; j++)
#pragma unroll
      for (int v = 0; v < 8; v++) {
        int gm = M0 + wm * 32 + i * 16 + v + 8 * hf;  // output row (channel)
        int gn = N0 + wn * 32 + j * 16 + nn;          // output col (pixel)
        float val = acc[i][j][v] + bias[gm];
        if (mode == 0) {
          int seg = gm >> 9, oc = gm & 511, hh = oc >> 6, dd = oc & 63;
          if (seg == 2) {
            vd[(((size_t)b * H_ + hh) * D_ + dd) * N_ + gn] = (bf16)val;
          } else {
            size_t qi = (((size_t)b * H_ + hh) * N_ + gn) * D_ + dd;
            if (seg == 0) qd[qi] = (bf16)val;
            else          kd[qi] = (bf16)val;
          }
        } else {
          size_t oi = ((size_t)b * C_ + gm) * N_ + gn;
          outd[oi] = xres[oi] + val;
        }
      }
}

// ---------------------------------------------------------------------------
// Kernel 4: flash attention. One block per (b, head, 128-query block).
// 8 waves; each wave owns 16 query rows. Streams 64-key K/V tiles through LDS
// (async copies), online softmax in f32, P transposed through a per-wave LDS
// tile for the P*V WMMA. Output -> o_t[b][n][c] bf16 (B^T layout for proj).
// ---------------------------------------------------------------------------
__global__ __launch_bounds__(256) void attn_flash(const bf16* __restrict__ q_t,
                                                  const bf16* __restrict__ k_t,
                                                  const bf16* __restrict__ v_t,
                                                  bf16* __restrict__ o_t) {
  __shared__ __align__(16) bf16 sK[64 * 64];      // [key][d]
  __shared__ __align__(16) bf16 sV[64 * 64];      // [d][key]
  __shared__ __align__(16) bf16 sP[8][16 * 64];   // per-wave P tile [16][64]
  int b = blockIdx.z, hh = blockIdx.y, q0 = blockIdx.x * 128;
  int tid = threadIdx.x, wave = tid >> 5, lane = tid & 31;
  int nn = lane & 15, hf = lane >> 4;
  const bf16* qb = q_t + ((size_t)(b * H_ + hh)) * N_ * D_;  // [n][64]
  const bf16* kb = k_t + ((size_t)(b * H_ + hh)) * N_ * D_;  // [n][64]
  const bf16* vb = v_t + ((size_t)(b * H_ + hh)) * D_ * N_;  // [64][n]
  int qrow = q0 + wave * 16;
  bf16x16 aq0 = frag_a(qb + (size_t)qrow * D_ + 0, D_);
  bf16x16 aq1 = frag_a(qb + (size_t)qrow * D_ + 32, D_);
  f32x8 z = {0.f, 0.f, 0.f, 0.f, 0.f, 0.f, 0.f, 0.f};
  f32x8 oA[4] = {z, z, z, z};
  float mrow[8], lrow[8];
#pragma unroll
  for (int v = 0; v < 8; v++) { mrow[v] = -INFINITY; lrow[v] = 0.f; }

  int row0 = (tid + 0)   >> 2, c40 = (tid + 0)   & 3;
  int row1 = (tid + 256) >> 2, c41 = (tid + 256) & 3;

  for (int j0 = 0; j0 < N_; j0 += 64) {
    copy16_to_lds(sK + row0 * 64 + c40 * 8,
                  kb + (size_t)(j0 + row0) * D_ + c40 * 8);
    copy16_to_lds(sK + row1 * 64 + c41 * 8,
                  kb + (size_t)(j0 + row1) * D_ + c41 * 8);
    copy16_to_lds(sV + row0 * 64 + c40 * 8,
                  vb + (size_t)row0 * N_ + j0 + c40 * 8);
    copy16_to_lds(sV + row1 * 64 + c41 * 8,
                  vb + (size_t)row1 * N_ + j0 + c41 * 8);
    wait_lds_copies();
    __syncthreads();

    // S = Q * K^T  (16 queries x 64 keys), scale d^-0.5 = 0.125
    f32x8 s[4] = {z, z, z, z};
#pragma unroll
    for (int t = 0; t < 4; t++) {
      s[t] = wmma_bf16(aq0, frag_b(sK + t * 16 * 64 + 0, 64), s[t]);
      s[t] = wmma_bf16(aq1, frag_b(sK + t * 16 * 64 + 32, 64), s[t]);
    }

    // online softmax: row r = v + 8*hf lives across lanes nn (16-lane groups)
#pragma unroll
    for (int v = 0; v < 8; v++) {
      float mx = -INFINITY;
#pragma unroll
      for (int t = 0; t < 4; t++) {
        s[t][v] *= 0.125f;
        mx = fmaxf(mx, s[t][v]);
      }
#pragma unroll
      for (int off = 8; off >= 1; off >>= 1)
        mx = fmaxf(mx, __shfl_xor(mx, off, 32));
      float mnew = fmaxf(mrow[v], mx);
      float corr = __expf(mrow[v] - mnew);
      float rs = 0.f;
#pragma unroll
      for (int t = 0; t < 4; t++) {
        float p = __expf(s[t][v] - mnew);
        s[t][v] = p;
        rs += p;
      }
#pragma unroll
      for (int off = 8; off >= 1; off >>= 1) rs += __shfl_xor(rs, off, 32);
      lrow[v] = lrow[v] * corr + rs;
      mrow[v] = mnew;
#pragma unroll
      for (int t = 0; t < 4; t++) oA[t][v] *= corr;
      int mr = v + 8 * hf;
#pragma unroll
      for (int t = 0; t < 4; t++)
        sP[wave][mr * 64 + t * 16 + nn] = (bf16)s[t][v];
    }

    // O += P * V  (A = P [16 q][64 key], B^T = sV [d][key])
    bf16x16 ap0 = frag_a(&sP[wave][0], 64);
    bf16x16 ap1 = frag_a(&sP[wave][32], 64);
#pragma unroll
    for (int t = 0; t < 4; t++) {
      oA[t] = wmma_bf16(ap0, frag_b(sV + t * 16 * 64 + 0, 64), oA[t]);
      oA[t] = wmma_bf16(ap1, frag_b(sV + t * 16 * 64 + 32, 64), oA[t]);
    }
    __syncthreads();
  }

#pragma unroll
  for (int v = 0; v < 8; v++) {
    float inv = 1.f / lrow[v];
    int p = q0 + wave * 16 + v + 8 * hf;
#pragma unroll
    for (int t = 0; t < 4; t++)
      o_t[((size_t)b * N_ + p) * C_ + hh * D_ + t * 16 + nn] =
          (bf16)(oA[t][v] * inv);
  }
}

// ---------------------------------------------------------------------------
extern "C" void kernel_launch(void* const* d_in, const int* in_sizes, int n_in,
                              void* d_out, int out_size, void* d_ws,
                              size_t ws_size, hipStream_t stream) {
  (void)in_sizes; (void)n_in; (void)out_size; (void)ws_size;
  const float* x      = (const float*)d_in[0];
  const float* w_qkv  = (const float*)d_in[1];
  const float* b_qkv  = (const float*)d_in[2];
  const float* w_proj = (const float*)d_in[3];
  const float* b_proj = (const float*)d_in[4];
  const float* gamma  = (const float*)d_in[5];
  const float* beta   = (const float*)d_in[6];
  float* out = (float*)d_out;

  char* ws = (char*)d_ws;
  size_t off = 0;
  auto alloc = [&](size_t elems) {
    bf16* p = (bf16*)(ws + off);
    off += elems * sizeof(bf16);
    off = (off + 255) & ~(size_t)255;
    return p;
  };
  bf16* wqb = alloc((size_t)3 * C_ * C_);
  bf16* wpb = alloc((size_t)C_ * C_);
  bf16* xn  = alloc((size_t)B_ * N_ * C_);
  bf16* qd  = alloc((size_t)B_ * H_ * N_ * D_);
  bf16* kd  = alloc((size_t)B_ * H_ * N_ * D_);
  bf16* vd  = alloc((size_t)B_ * H_ * D_ * N_);
  bf16* ot  = alloc((size_t)B_ * N_ * C_);

  cvt_weights<<<4096, 256, 0, stream>>>(w_qkv, w_proj, wqb, wpb);
  groupnorm_kernel<<<B_ * 32, 256, 0, stream>>>(x, gamma, beta, xn);
  gemm_wmma<<<dim3(N_ / 64, (3 * C_) / 128, B_), 256, 0, stream>>>(
      wqb, xn, b_qkv, qd, kd, vd, nullptr, nullptr, 0);
  attn_flash<<<dim3(N_ / 128, H_, B_), 256, 0, stream>>>(qd, kd, vd, ot);
  gemm_wmma<<<dim3(N_ / 64, C_ / 128, B_), 256, 0, stream>>>(
      wpb, ot, b_proj, nullptr, nullptr, nullptr, out, x, 1);
}
